// SetConv_26353919328870
// MI455X (gfx1250) — matrix operand order repaired and checked by reading, not compiled, not run on hardware
//
#include <hip/hip_runtime.h>

typedef __attribute__((ext_vector_type(16))) _Float16     v16h;
typedef __attribute__((ext_vector_type(8)))  float        v8f;
typedef __attribute__((ext_vector_type(4)))  unsigned int v4u;
typedef __attribute__((ext_vector_type(8)))  int          v8i_t;
typedef __attribute__((ext_vector_type(4)))  int          v4i_t;

#define B_     4
#define NQ_    4096
#define NC_    4096
#define DY_    8
#define DOUT_  (DY_ + 1)
#define WPB    8                 // waves per block
#define STAGE  128               // contexts per TDM stage
#define NSTAGE (NC_ / STAGE)     // 32 stages

// ---- Tensor Data Mover: 1-row tile (contiguous run) global -> LDS ----
// D# layout per CDNA5 ISA ch.8 (group0 128b, group1 256b); 6-arg builtin form
// (clang-23 / therock-10.0 headers): (v4u, v8i, v4i, v4i, v8i, i32 cpol).
__device__ __forceinline__ void tdm_load_1d(unsigned lds_addr, const float* gptr,
                                            unsigned tile_elems, unsigned tensor_elems)
{
    unsigned long long ga = (unsigned long long)(uintptr_t)gptr;
    v4u g0;
    g0[0] = 1u;                                       // count=1, user descriptor
    g0[1] = lds_addr;                                 // lds_addr (bytes)
    g0[2] = (unsigned)ga;                             // global_addr[31:0]
    g0[3] = (unsigned)((ga >> 32) & 0x01ffffffu)      // global_addr[56:32]
          | (2u << 30);                               // type = 2
    v8i_t g1;
    g1[0] = (int)(2u << 16);                          // data_size=2 (4B), wg_mask=0
    g1[1] = (int)((tensor_elems & 0xffffu) << 16);    // tensor_dim0[15:0] @ [63:48]
    g1[2] = (int)(((tensor_elems >> 16) & 0xffffu)    // tensor_dim0[31:16] @ [79:64]
          | (1u << 16));                              // tensor_dim1 = 1 @ [95:80]
    g1[3] = (int)((tile_elems & 0xffffu) << 16);      // tile_dim0 @ [127:112]
    g1[4] = 1;                                        // tile_dim1=1, tile_dim2=0
    g1[5] = (int)tensor_elems;                        // tensor_dim0_stride[31:0]
    g1[6] = 0;
    g1[7] = 0;
    v4i_t z4 = {0, 0, 0, 0};
    v8i_t z8 = {0, 0, 0, 0, 0, 0, 0, 0};
    __builtin_amdgcn_tensor_load_to_lds(g0, g1, z4, z4, z8, /*cpol=*/0);
}

__device__ __forceinline__ unsigned lds_off(const void* p)
{   // flat LDS address: low 32 bits are the LDS byte offset
    return (unsigned)(uintptr_t)p;
}

// w = exp(-d2/(2*ls^2)) = exp2(d2 * c2),  c2 = -log2(e)/(2*ls^2)
__global__ __launch_bounds__(WPB * 32)
void setconv_wmma(const float* __restrict__ xq_g,
                  const float* __restrict__ xc_g,
                  const float* __restrict__ yc_g,
                  const float* __restrict__ lls_g,
                  float* __restrict__ out)
{
    __shared__ float s_xc[2][STAGE * 2];    // (c, 2) f32, double buffered
    __shared__ float s_yc[2][STAGE * DY_];  // (c, 8) f32, double buffered

    const int lane = threadIdx.x & 31;
    const int wave = threadIdx.x >> 5;
    const int b    = blockIdx.x / (NQ_ / 16 / WPB);          // uniform per block
    const int tile = blockIdx.x * WPB + wave;
    const int q0   = (tile % (NQ_ / 16)) * 16;

    const int m    = lane & 15;  // A row (query) == D column N
    const int half = lane >> 4;

    const float LOG2E  = 1.44269504088896340736f;
    const float lls    = lls_g[0];
    const float inv2l2 = 0.5f * __builtin_amdgcn_exp2f(-2.0f * lls * LOG2E);
    const float c2     = -inv2l2 * LOG2E;

    const float2 xq = ((const float2*)xq_g)[(size_t)b * NQ_ + q0 + m];

    const float* xc = xc_g + (size_t)b * NC_ * 2;
    const float* yc = yc_g + (size_t)b * NC_ * DY_;

    // Prologue: TDM-load stage 0 into buffer 0 (one wave drives the DMA engine).
    if (wave == 0) {
        tdm_load_1d(lds_off(&s_xc[0][0]), xc, STAGE * 2,  NC_ * 2);
        tdm_load_1d(lds_off(&s_yc[0][0]), yc, STAGE * DY_, NC_ * DY_);
    }

    v8f acc = {};

    for (int s = 0; s < NSTAGE; ++s) {
        const int buf = s & 1;
        if (wave == 0) __builtin_amdgcn_s_wait_tensorcnt(0);
        __syncthreads();                      // publish stage s LDS tile

        // Prefetch stage s+1 into the other buffer (safe: all waves passed the
        // barrier, so nobody still reads buf^1 from stage s-1).
        if (wave == 0 && s + 1 < NSTAGE) {
            const int nb = buf ^ 1;
            tdm_load_1d(lds_off(&s_xc[nb][0]), xc + (size_t)(s + 1) * STAGE * 2,
                        STAGE * 2,  NC_ * 2);
            tdm_load_1d(lds_off(&s_yc[nb][0]), yc + (size_t)(s + 1) * STAGE * DY_,
                        STAGE * DY_, NC_ * DY_);
        }

        const float2* xcb = (const float2*)&s_xc[buf][0];
        const float*  ycb = &s_yc[buf][0];

#pragma unroll
        for (int ch = 0; ch < STAGE / 32; ++ch) {
            const int cbase = ch * 32;

            // ---- A fragment: RBF weights, 16x32 f16 (ISA 16-bit A layout) ----
            v16h a;
#pragma unroll
            for (int e = 0; e < 16; ++e) {
                const int v   = e >> 1;
                const int pos = e & 1;
                const int kk  = (v < 4) ? (half * 8 + 2 * v + pos)
                                        : (16 + half * 8 + 2 * (v - 4) + pos);
                const float2 p  = xcb[cbase + kk];
                const float  dx = xq.x - p.x;
                const float  dy = xq.y - p.y;
                const float  d2 = dx * dx + dy * dy;
                a[e] = (_Float16)__builtin_amdgcn_exp2f(d2 * c2);
            }

            // ---- B fragment: y 32x16 f16 (cols 0..7 = y, col 8 = ones) ----
            v16h bf;
#pragma unroll
            for (int e = 0; e < 16; ++e) {
                const int kk = (e < 8) ? (half * 8 + e) : (16 + half * 8 + (e - 8));
                float val;
                if (m < DY_)       val = ycb[(cbase + kk) * DY_ + m];
                else if (m == DY_) val = 1.0f;   // density column
                else               val = 0.0f;   // padding
                bf[e] = (_Float16)val;
            }

            acc = __builtin_amdgcn_wmma_f32_16x16x32_f16(
                false, a, false, bf, (short)0, acc, false, false);
        }
    }

    // Epilogue: acc[r] = D[half*8 + r, N=m]; column 8 holds density.
#pragma unroll
    for (int r = 0; r < 8; ++r) {
        const float dens = __shfl(acc[r], (lane & 16) | 8, 32);
        if (m <= DY_) {
            const int   row = q0 + half * 8 + r;
            const float val = (m < DY_) ? acc[r] / (dens + 1e-8f) : dens;
            out[((size_t)b * NQ_ + row) * DOUT_ + m] = val;
        }
    }
}

extern "C" void kernel_launch(void* const* d_in, const int* in_sizes, int n_in,
                              void* d_out, int out_size, void* d_ws, size_t ws_size,
                              hipStream_t stream)
{
    const float* xq  = (const float*)d_in[0];  // (B, NQ, 2)
    const float* xc  = (const float*)d_in[1];  // (B, NC, 2)
    const float* yc  = (const float*)d_in[2];  // (B, NC, 8)
    const float* lls = (const float*)d_in[3];  // scalar
    float*       out = (float*)d_out;          // (B, NQ, 9)

    const int tiles  = B_ * NQ_ / 16;          // 1024 query tiles
    const int blocks = tiles / WPB;            // 128 blocks x 8 waves
    setconv_wmma<<<blocks, WPB * 32, 0, stream>>>(xq, xc, yc, lls, out);
}
